// LSTM_fusion_70085276336622
// MI455X (gfx1250) — matrix operand order repaired and checked by reading, not compiled, not run on hardware
//
#include <hip/hip_runtime.h>

#define B_ 4096
#define L_ 32
#define FEAT_ 128
#define FM_ 64
#define H_ 512
#define C_ 256

typedef __attribute__((ext_vector_type(16))) __bf16 bf16x16;
typedef __attribute__((ext_vector_type(8)))  __bf16 bf16x8;
typedef __attribute__((ext_vector_type(8)))  float  f32x8;

__device__ __forceinline__ unsigned short f2bf(float x) {
    union { float f; unsigned u; } v; v.f = x;
    unsigned r = v.u + 0x7FFFu + ((v.u >> 16) & 1u);
    return (unsigned short)(r >> 16);
}

__device__ __forceinline__ float sigmoidf_(float x) {
    return 1.0f / (1.0f + __expf(-x));
}

union frag_t { bf16x16 v; bf16x8 h[2]; };

// ---------------------------------------------------------------------------
// Generic bf16 WMMA GEMM: D[M,N] = A[M,K] * W[N,K]^T (+bias) (act)
// Block 256 thr = 8 waves (2M x 4N); wave tile 32x64; block tile 64x256.
// grid = (N/256, M/64). act: 0=none 1=relu 2=sigmoid. Dh!=null -> bf16 out.
// ---------------------------------------------------------------------------
__global__ __launch_bounds__(256) void gemm_bf16_wmma(
    const __bf16* __restrict__ A, int lda,
    const __bf16* __restrict__ W, int K,
    const float* __restrict__ bias,
    float* __restrict__ Df,
    unsigned short* __restrict__ Dh,
    int ldd, int act)
{
    const int lane  = threadIdx.x & 31;
    const int wave  = threadIdx.x >> 5;
    const int wm    = wave >> 2;
    const int wn    = wave & 3;
    const int Mw    = blockIdx.y * 64 + wm * 32;
    const int Nw    = blockIdx.x * 256 + wn * 64;
    const int lhalf = lane >> 4;
    const int l15   = lane & 15;

    f32x8 accv[2][4];
#pragma unroll
    for (int mi = 0; mi < 2; ++mi)
#pragma unroll
        for (int ni = 0; ni < 4; ++ni)
#pragma unroll
            for (int r = 0; r < 8; ++r) accv[mi][ni][r] = 0.0f;

    for (int k = 0; k < K; k += 32) {
        frag_t a[2], b[4];
#pragma unroll
        for (int mi = 0; mi < 2; ++mi) {
            const __bf16* p = A + (size_t)(Mw + mi * 16 + l15) * lda + k + (lhalf << 3);
            a[mi].h[0] = *(const bf16x8*)(p);
            a[mi].h[1] = *(const bf16x8*)(p + 16);
        }
#pragma unroll
        for (int ni = 0; ni < 4; ++ni) {
            const __bf16* p = W + (size_t)(Nw + ni * 16 + l15) * K + k + (lhalf << 3);
            b[ni].h[0] = *(const bf16x8*)(p);
            b[ni].h[1] = *(const bf16x8*)(p + 16);
        }
#pragma unroll
        for (int mi = 0; mi < 2; ++mi)
#pragma unroll
            for (int ni = 0; ni < 4; ++ni)
                accv[mi][ni] = __builtin_amdgcn_wmma_f32_16x16x32_bf16(
                    false, a[mi].v, false, b[ni].v,
                    (short)0, accv[mi][ni], false, false);
    }

#pragma unroll
    for (int mi = 0; mi < 2; ++mi) {
#pragma unroll
        for (int r = 0; r < 8; ++r) {
            const int row = Mw + mi * 16 + r + (lhalf << 3);
#pragma unroll
            for (int ni = 0; ni < 4; ++ni) {
                const int col = Nw + ni * 16 + l15;
                float v = accv[mi][ni][r];
                if (bias) v += bias[col];
                size_t o = (size_t)row * ldd + col;
                if (act == 1)      v = v > 0.0f ? v : 0.0f;
                else if (act == 2) v = sigmoidf_(v);
                if (Dh) Dh[o] = f2bf(v);
                else    Df[o] = v;
            }
        }
    }
}

// ---------------------------------------------------------------------------
// Fused LSTM step: one launch computes
//   gates = biasP + A1 * W1^T + A2 * W2^T   (W1/W2 rows gate-permuted)
//   i,f,g,o = act(gates);  c = f*c + i*g;  h_next = o*tanh(c)   (in epilogue)
// Permuted column layout: within each 64-col group: [i(16) f(16) g(16) o(16)]
// of the same 16 hidden units, so a wave's 4 N-subtiles are exactly i/f/g/o
// at matching lane/row positions -> in-register cell update, no gates buffer.
// h double-buffered (reads A2=h_prev, writes hb=h_next); c updated in place.
// grid = (4H/256 = 8, B/64 = 64).
// ---------------------------------------------------------------------------
__global__ __launch_bounds__(256) void lstm_step_wmma(
    const __bf16* __restrict__ A1, int lda1, int K1, const __bf16* __restrict__ W1,
    const __bf16* __restrict__ A2, int lda2, int K2, const __bf16* __restrict__ W2,
    const float* __restrict__ biasP,
    float* __restrict__ c, unsigned short* __restrict__ hb)
{
    const int lane  = threadIdx.x & 31;
    const int wave  = threadIdx.x >> 5;
    const int wm    = wave >> 2;
    const int wn    = wave & 3;
    const int Mw    = blockIdx.y * 64 + wm * 32;
    const int Nw    = blockIdx.x * 256 + wn * 64;
    const int lhalf = lane >> 4;
    const int l15   = lane & 15;

    f32x8 accv[2][4];
#pragma unroll
    for (int mi = 0; mi < 2; ++mi)
#pragma unroll
        for (int ni = 0; ni < 4; ++ni)
#pragma unroll
            for (int r = 0; r < 8; ++r) accv[mi][ni][r] = 0.0f;

    // ---- pass 1: A1 * W1^T ----
    for (int k = 0; k < K1; k += 32) {
        frag_t a[2], b[4];
#pragma unroll
        for (int mi = 0; mi < 2; ++mi) {
            const __bf16* p = A1 + (size_t)(Mw + mi * 16 + l15) * lda1 + k + (lhalf << 3);
            a[mi].h[0] = *(const bf16x8*)(p);
            a[mi].h[1] = *(const bf16x8*)(p + 16);
        }
#pragma unroll
        for (int ni = 0; ni < 4; ++ni) {
            const __bf16* p = W1 + (size_t)(Nw + ni * 16 + l15) * K1 + k + (lhalf << 3);
            b[ni].h[0] = *(const bf16x8*)(p);
            b[ni].h[1] = *(const bf16x8*)(p + 16);
        }
#pragma unroll
        for (int mi = 0; mi < 2; ++mi)
#pragma unroll
            for (int ni = 0; ni < 4; ++ni)
                accv[mi][ni] = __builtin_amdgcn_wmma_f32_16x16x32_bf16(
                    false, a[mi].v, false, b[ni].v,
                    (short)0, accv[mi][ni], false, false);
    }

    // ---- pass 2: + A2 * W2^T ----
    for (int k = 0; k < K2; k += 32) {
        frag_t a[2], b[4];
#pragma unroll
        for (int mi = 0; mi < 2; ++mi) {
            const __bf16* p = A2 + (size_t)(Mw + mi * 16 + l15) * lda2 + k + (lhalf << 3);
            a[mi].h[0] = *(const bf16x8*)(p);
            a[mi].h[1] = *(const bf16x8*)(p + 16);
        }
#pragma unroll
        for (int ni = 0; ni < 4; ++ni) {
            const __bf16* p = W2 + (size_t)(Nw + ni * 16 + l15) * K2 + k + (lhalf << 3);
            b[ni].h[0] = *(const bf16x8*)(p);
            b[ni].h[1] = *(const bf16x8*)(p + 16);
        }
#pragma unroll
        for (int mi = 0; mi < 2; ++mi)
#pragma unroll
            for (int ni = 0; ni < 4; ++ni)
                accv[mi][ni] = __builtin_amdgcn_wmma_f32_16x16x32_bf16(
                    false, a[mi].v, false, b[ni].v,
                    (short)0, accv[mi][ni], false, false);
    }

    // ---- fused cell epilogue ----
    const float bi = biasP[Nw +  0 + l15];
    const float bf = biasP[Nw + 16 + l15];
    const float bg = biasP[Nw + 32 + l15];
    const float bo = biasP[Nw + 48 + l15];
    const int   u  = ((Nw >> 6) << 4) + l15;     // hidden unit index

#pragma unroll
    for (int mi = 0; mi < 2; ++mi) {
#pragma unroll
        for (int r = 0; r < 8; ++r) {
            const int row = Mw + mi * 16 + r + (lhalf << 3);
            float ig = sigmoidf_(accv[mi][0][r] + bi);
            float fg = sigmoidf_(accv[mi][1][r] + bf);
            float gg = tanhf(accv[mi][2][r] + bg);
            float og = sigmoidf_(accv[mi][3][r] + bo);
            size_t idx = (size_t)row * H_ + u;
            float cn = fg * c[idx] + ig * gg;
            c[idx] = cn;
            hb[idx] = f2bf(og * tanhf(cn));
        }
    }
}

// ---------------------------------------------------------------------------
// Elementwise helpers
// ---------------------------------------------------------------------------
__global__ __launch_bounds__(256) void cast_f32_bf16(
    const float* __restrict__ in, unsigned short* __restrict__ out, size_t n)
{
    size_t i = (size_t)blockIdx.x * blockDim.x + threadIdx.x;
    size_t stride = (size_t)gridDim.x * blockDim.x;
    for (; i < n; i += stride) out[i] = f2bf(in[i]);
}

// Gate-permuted weight cast: new row n' -> orig row gate*H + unit,
// where gate = (n'>>4)&3, unit = ((n'>>6)<<4) | (n'&15).
__global__ __launch_bounds__(256) void permute_cast_lstm(
    const float* __restrict__ src, unsigned short* __restrict__ dst, int K)
{
    size_t n = (size_t)4 * H_ * K;
    size_t i = (size_t)blockIdx.x * blockDim.x + threadIdx.x;
    size_t stride = (size_t)gridDim.x * blockDim.x;
    for (; i < n; i += stride) {
        int np = (int)(i / K);
        int k  = (int)(i - (size_t)np * K);
        int gate = (np >> 4) & 3;
        int unit = ((np >> 6) << 4) | (np & 15);
        int orig = gate * H_ + unit;
        dst[i] = f2bf(src[(size_t)orig * K + k]);
    }
}

__global__ __launch_bounds__(256) void permute_bias_pair(
    const float* __restrict__ a, const float* __restrict__ b,
    float* __restrict__ out)
{
    int np = blockIdx.x * blockDim.x + threadIdx.x;   // [0, 4H)
    int gate = (np >> 4) & 3;
    int unit = ((np >> 6) << 4) | (np & 15);
    int orig = gate * H_ + unit;
    out[np] = a[orig] + b[orig];
}

__global__ __launch_bounds__(256) void zero_u32(unsigned* __restrict__ p, size_t n)
{
    size_t i = (size_t)blockIdx.x * blockDim.x + threadIdx.x;
    size_t stride = (size_t)gridDim.x * blockDim.x;
    for (; i < n; i += stride) p[i] = 0u;
}

// Top-8 wideband selection; one wave per row.
__global__ __launch_bounds__(256) void topk_mask(
    const float* __restrict__ out_sub, const float* __restrict__ y,
    unsigned short* __restrict__ inmm)
{
    int lane = threadIdx.x & 31;
    int row  = blockIdx.x * 8 + (threadIdx.x >> 5);
    const float* os = out_sub + (size_t)row * C_ + lane * 8;
    float s = 0.0f;
#pragma unroll
    for (int q = 0; q < 8; ++q) s += os[q];

    float v = s;
    int keep = 0;
#pragma unroll
    for (int it = 0; it < 8; ++it) {
        float m = v; int ml = lane;
#pragma unroll
        for (int off = 16; off > 0; off >>= 1) {
            float ov = __shfl_xor(m, off, 32);
            int   ol = __shfl_xor(ml, off, 32);
            if (ov > m || (ov == m && ol < ml)) { m = ov; ml = ol; }
        }
        if (lane == ml) { keep = 1; v = -__builtin_inff(); }
    }
    float m0 = keep ? y[(size_t)row * FM_ + 2 * lane]     : 0.0f;
    float m1 = keep ? y[(size_t)row * FM_ + 2 * lane + 1] : 0.0f;
    inmm[(size_t)row * FM_ + 2 * lane]     = f2bf(m0);
    inmm[(size_t)row * FM_ + 2 * lane + 1] = f2bf(m1);
}

__global__ __launch_bounds__(256) void concat_bf16(
    const float* __restrict__ out_sub, const float* __restrict__ out_mm,
    unsigned short* __restrict__ catb)
{
    int idx = blockIdx.x * blockDim.x + threadIdx.x;
    int b = idx >> 9;
    int j = idx & (2 * C_ - 1);
    float v = (j < C_) ? out_sub[(size_t)b * C_ + j]
                       : out_mm[(size_t)b * C_ + j - C_];
    catb[idx] = f2bf(v);
}

__global__ __launch_bounds__(256) void gated_fuse(
    const float* __restrict__ out_sub, const float* __restrict__ out_mm,
    const float* __restrict__ gate, unsigned short* __restrict__ fusedb)
{
    int idx = blockIdx.x * blockDim.x + threadIdx.x;
    int b = idx >> 9;
    int j = idx & (2 * C_ - 1);
    float cv = (j < C_) ? out_sub[(size_t)b * C_ + j]
                        : out_mm[(size_t)b * C_ + j - C_];
    fusedb[idx] = f2bf(cv * gate[idx]);
}

// ---------------------------------------------------------------------------
extern "C" void kernel_launch(void* const* d_in, const int* in_sizes, int n_in,
                              void* d_out, int out_size, void* d_ws, size_t ws_size,
                              hipStream_t stream)
{
    (void)in_sizes; (void)n_in; (void)out_size; (void)ws_size;

    const float* x    = (const float*)d_in[0];
    const float* y    = (const float*)d_in[1];
    const float* Wih0 = (const float*)d_in[2];
    const float* Whh0 = (const float*)d_in[3];
    const float* bih0 = (const float*)d_in[4];
    const float* bhh0 = (const float*)d_in[5];
    const float* Wih1 = (const float*)d_in[6];
    const float* Whh1 = (const float*)d_in[7];
    const float* bih1 = (const float*)d_in[8];
    const float* bhh1 = (const float*)d_in[9];
    const float* decW = (const float*)d_in[10];
    const float* decb = (const float*)d_in[11];
    const float* mmW1 = (const float*)d_in[12];
    const float* mmb1 = (const float*)d_in[13];
    const float* mmW2 = (const float*)d_in[14];
    const float* mmb2 = (const float*)d_in[15];
    const float* mmW3 = (const float*)d_in[16];
    const float* mmb3 = (const float*)d_in[17];
    const float* attW = (const float*)d_in[18];
    const float* attb = (const float*)d_in[19];
    const float* fuW1 = (const float*)d_in[20];
    const float* fub1 = (const float*)d_in[21];
    const float* fuW2 = (const float*)d_in[22];
    const float* fub2 = (const float*)d_in[23];
    const float* fuW3 = (const float*)d_in[24];
    const float* fub3 = (const float*)d_in[25];

    float* out_sub = (float*)d_out;
    float* out_mm  = out_sub + (size_t)B_ * C_;
    float* out_fu  = out_mm  + (size_t)B_ * C_;

    char* wsb = (char*)d_ws;
    size_t off = 0;
    auto alloc = [&](size_t bytes) -> char* {
        off = (off + 255) & ~(size_t)255;
        char* p = wsb + off;
        off += bytes;
        return p;
    };

    unsigned short* xb     = (unsigned short*)alloc((size_t)B_ * L_ * FEAT_ * 2);
    unsigned short* Wih0p  = (unsigned short*)alloc((size_t)4 * H_ * FEAT_ * 2);
    unsigned short* Whh0p  = (unsigned short*)alloc((size_t)4 * H_ * H_ * 2);
    unsigned short* Wih1p  = (unsigned short*)alloc((size_t)4 * H_ * H_ * 2);
    unsigned short* Whh1p  = (unsigned short*)alloc((size_t)4 * H_ * H_ * 2);
    unsigned short* decWb  = (unsigned short*)alloc((size_t)C_ * H_ * 2);
    unsigned short* mmW1b  = (unsigned short*)alloc((size_t)H_ * FM_ * 2);
    unsigned short* mmW2b  = (unsigned short*)alloc((size_t)H_ * H_ * 2);
    unsigned short* mmW3b  = (unsigned short*)alloc((size_t)C_ * H_ * 2);
    unsigned short* attWb  = (unsigned short*)alloc((size_t)2 * C_ * 2 * C_ * 2);
    unsigned short* fuW1b  = (unsigned short*)alloc((size_t)H_ * 2 * C_ * 2);
    unsigned short* fuW2b  = (unsigned short*)alloc((size_t)H_ * H_ * 2);
    unsigned short* fuW3b  = (unsigned short*)alloc((size_t)C_ * H_ * 2);
    float* combP0          = (float*)alloc(4 * H_ * 4);
    float* combP1          = (float*)alloc(4 * H_ * 4);
    // state block (zeroed together): h0 ping/pong, h1 ping/pong, c0, c1
    unsigned short* h0p0   = (unsigned short*)alloc((size_t)B_ * H_ * 2);
    unsigned short* h0p1   = (unsigned short*)alloc((size_t)B_ * H_ * 2);
    unsigned short* h1p0   = (unsigned short*)alloc((size_t)B_ * H_ * 2);
    unsigned short* h1p1   = (unsigned short*)alloc((size_t)B_ * H_ * 2);
    float* c0              = (float*)alloc((size_t)B_ * H_ * 4);
    float* c1              = (float*)alloc((size_t)B_ * H_ * 4);
    unsigned short* inmmb  = (unsigned short*)alloc((size_t)B_ * FM_ * 2);
    unsigned short* mh1b   = (unsigned short*)alloc((size_t)B_ * H_ * 2);
    unsigned short* mh2b   = (unsigned short*)alloc((size_t)B_ * H_ * 2);
    unsigned short* catb   = (unsigned short*)alloc((size_t)B_ * 2 * C_ * 2);
    float* gatef           = (float*)alloc((size_t)B_ * 2 * C_ * 4);
    unsigned short* fusedb = (unsigned short*)alloc((size_t)B_ * 2 * C_ * 2);
    unsigned short* fh1b   = (unsigned short*)alloc((size_t)B_ * H_ * 2);
    unsigned short* fh2b   = (unsigned short*)alloc((size_t)B_ * H_ * 2);

    auto cast = [&](const float* src, unsigned short* dst, size_t n) {
        int blocks = (int)((n + 255) / 256);
        if (blocks > 2048) blocks = 2048;
        cast_f32_bf16<<<blocks, 256, 0, stream>>>(src, dst, n);
    };
    cast(x, xb, (size_t)B_ * L_ * FEAT_);
    cast(decW, decWb, (size_t)C_ * H_);
    cast(mmW1, mmW1b, (size_t)H_ * FM_);
    cast(mmW2, mmW2b, (size_t)H_ * H_);
    cast(mmW3, mmW3b, (size_t)C_ * H_);
    cast(attW, attWb, (size_t)2 * C_ * 2 * C_);
    cast(fuW1, fuW1b, (size_t)H_ * 2 * C_);
    cast(fuW2, fuW2b, (size_t)H_ * H_);
    cast(fuW3, fuW3b, (size_t)C_ * H_);

    permute_cast_lstm<<<2048, 256, 0, stream>>>(Wih0, Wih0p, FEAT_);
    permute_cast_lstm<<<2048, 256, 0, stream>>>(Whh0, Whh0p, H_);
    permute_cast_lstm<<<2048, 256, 0, stream>>>(Wih1, Wih1p, H_);
    permute_cast_lstm<<<2048, 256, 0, stream>>>(Whh1, Whh1p, H_);
    permute_bias_pair<<<(4 * H_) / 256, 256, 0, stream>>>(bih0, bhh0, combP0);
    permute_bias_pair<<<(4 * H_) / 256, 256, 0, stream>>>(bih1, bhh1, combP1);

    // zero h ping-pongs + c (contiguous): 4*(B*H*2) + 2*(B*H*4) bytes
    {
        size_t zbytes = (size_t)4 * B_ * H_ * 2 + (size_t)2 * B_ * H_ * 4;
        zero_u32<<<2048, 256, 0, stream>>>((unsigned*)h0p0, zbytes / 4);
    }

    auto gemm = [&](const unsigned short* A, int lda, const unsigned short* Wt, int K,
                    const float* bias, float* Df, unsigned short* Dh,
                    int ldd, int N, int act) {
        dim3 grid(N / 256, B_ / 64);
        gemm_bf16_wmma<<<grid, 256, 0, stream>>>(
            (const __bf16*)A, lda, (const __bf16*)Wt, K,
            bias, Df, Dh, ldd, act);
    };

    // ----- 2-layer LSTM scan: 2 fused launches per step -----
    unsigned short* h0buf[2] = { h0p0, h0p1 };
    unsigned short* h1buf[2] = { h1p0, h1p1 };
    int cur0 = 0, cur1 = 0;
    dim3 sgrid((4 * H_) / 256, B_ / 64);
    for (int t = 0; t < L_; ++t) {
        lstm_step_wmma<<<sgrid, 256, 0, stream>>>(
            (const __bf16*)(xb + (size_t)t * FEAT_), L_ * FEAT_, FEAT_, (const __bf16*)Wih0p,
            (const __bf16*)h0buf[cur0], H_, H_, (const __bf16*)Whh0p,
            combP0, c0, h0buf[cur0 ^ 1]);
        cur0 ^= 1;
        lstm_step_wmma<<<sgrid, 256, 0, stream>>>(
            (const __bf16*)h0buf[cur0], H_, H_, (const __bf16*)Wih1p,
            (const __bf16*)h1buf[cur1], H_, H_, (const __bf16*)Whh1p,
            combP1, c1, h1buf[cur1 ^ 1]);
        cur1 ^= 1;
    }
    unsigned short* h1fin = h1buf[cur1];

    // ----- decoder -----
    gemm(h1fin, H_, decWb, H_, decb, out_sub, nullptr, C_, C_, 0);

    // ----- top-8 wideband mask + input_mm -----
    topk_mask<<<B_ / 8, 256, 0, stream>>>(out_sub, y, inmmb);

    // ----- mm MLP branch -----
    gemm(inmmb, FM_, mmW1b, FM_, mmb1, nullptr, mh1b, H_, H_, 1);
    gemm(mh1b, H_, mmW2b, H_, mmb2, nullptr, mh2b, H_, H_, 1);
    gemm(mh2b, H_, mmW3b, H_, mmb3, out_mm, nullptr, C_, C_, 0);

    // ----- gated fusion -----
    concat_bf16<<<(B_ * 2 * C_) / 256, 256, 0, stream>>>(out_sub, out_mm, catb);
    gemm(catb, 2 * C_, attWb, 2 * C_, attb, gatef, nullptr, 2 * C_, 2 * C_, 2);
    gated_fuse<<<(B_ * 2 * C_) / 256, 256, 0, stream>>>(out_sub, out_mm, gatef, fusedb);

    gemm(fusedb, 2 * C_, fuW1b, 2 * C_, fub1, nullptr, fh1b, H_, H_, 1);
    gemm(fh1b, H_, fuW2b, H_, fub2, nullptr, fh2b, H_, H_, 1);
    gemm(fh2b, H_, fuW3b, H_, fub3, out_fu, nullptr, C_, C_, 0);
}
